// RingAttention_42752104464773
// MI455X (gfx1250) — matrix operand order repaired and checked by reading, not compiled, not run on hardware
//
#include <hip/hip_runtime.h>
#include <stdint.h>

// ---------------------------------------------------------------------------
// CDNA5 (gfx1250) attention block: RMSNorm -> QKV GEMM -> rotary -> flash
// attention -> out-proj GEMM.  Matmuls: v_wmma_f32_16x16x32_bf16 (wave32)
// with A-operand reuse hints; double-buffered LDS tiles filled by
// global_load_async_to_lds_b128 (ASYNCcnt) overlapped with WMMA compute;
// B-fragments software-pipelined one step ahead of their WMMA.
// ---------------------------------------------------------------------------

#define SEQ      4096
#define DIM      2048
#define HEADS    16
#define DHEAD    128
#define NQKV     (3 * DIM)        // 6144

typedef __bf16 bf16_t;
typedef __attribute__((ext_vector_type(16))) __bf16 v16bf;
typedef __attribute__((ext_vector_type(8)))  float  v8f;

union FragAB {            // 16 bf16 = 32 bytes = two 16B b128 loads
  v16bf v;
  uint4 q[2];
};

// RA: reuse-A hint; legal when consecutive WMMAs share the A operand.
template <bool RA>
static __device__ __forceinline__ v8f wmma_bf16(const v16bf& a, const v16bf& b, v8f c) {
  return __builtin_amdgcn_wmma_f32_16x16x32_bf16(false, a, false, b, (short)0, c,
                                                 RA, false);
}

// --- async global->LDS copy (16B per lane), ASYNCcnt-tracked ---------------
static __device__ __forceinline__ void async_ld16(bf16_t* lds, const bf16_t* g) {
  asm volatile("global_load_async_to_lds_b128 %0, %1, off"
               :: "v"((uint32_t)(uintptr_t)lds), "v"((uint64_t)(uintptr_t)g)
               : "memory");
}

static __device__ __forceinline__ void wait_async0() {
#if __has_builtin(__builtin_amdgcn_s_wait_asynccnt)
  __builtin_amdgcn_s_wait_asynccnt(0);
#else
  asm volatile("s_wait_asynccnt 0x0" ::: "memory");
#endif
}

static __device__ __forceinline__ void load_frag(FragAB& f, const bf16_t* p, int delta) {
  f.q[0] = *(const uint4*)(p);
  f.q[1] = *(const uint4*)(p + delta);
}

// ---------------------------------------------------------------------------
// 1) RMSNorm: xn = x / max(||x||,1e-12) * sqrt(DIM) * gamma   (f32 -> bf16)
// ---------------------------------------------------------------------------
__global__ void __launch_bounds__(256)
rmsnorm_kernel(const float* __restrict__ x, const float* __restrict__ gamma,
               bf16_t* __restrict__ xn) {
  const int row = blockIdx.x;
  const float* xr = x + (size_t)row * DIM;
  __shared__ float red[8];
  float ss = 0.f;
  for (int i = threadIdx.x; i < DIM; i += 256) { float v = xr[i]; ss += v * v; }
  for (int m = 16; m >= 1; m >>= 1) ss += __shfl_xor(ss, m, 32);
  if ((threadIdx.x & 31) == 0) red[threadIdx.x >> 5] = ss;
  __syncthreads();
  float tot = 0.f;
  #pragma unroll
  for (int i = 0; i < 8; ++i) tot += red[i];
  const float scale = 45.254833995939045f /*sqrt(2048)*/ / fmaxf(sqrtf(tot), 1e-12f);
  bf16_t* dst = xn + (size_t)row * DIM;
  for (int i = threadIdx.x; i < DIM; i += 256)
    dst[i] = (bf16_t)(xr[i] * scale * gamma[i]);
}

// ---------------------------------------------------------------------------
// 2) Transpose + cast f32 [K,N] -> bf16 [N,K] (WMMA B-operand layout)
// ---------------------------------------------------------------------------
__global__ void __launch_bounds__(256)
transpose_cast_kernel(const float* __restrict__ w, bf16_t* __restrict__ wt,
                      int K, int N) {
  __shared__ float tile[32][33];
  const int k0 = blockIdx.y * 32, n0 = blockIdx.x * 32;
  const int tx = threadIdx.x & 31, ty = threadIdx.x >> 5;   // ty: 0..7
  #pragma unroll
  for (int i = ty; i < 32; i += 8)
    tile[i][tx] = w[(size_t)(k0 + i) * N + (n0 + tx)];
  __syncthreads();
  #pragma unroll
  for (int i = ty; i < 32; i += 8)
    wt[(size_t)(n0 + i) * K + (k0 + tx)] = (bf16_t)tile[tx][i];
}

// ---------------------------------------------------------------------------
// 3) WMMA GEMM:  C[M,N] = A[M,K](bf16, row-major) * Bt[N,K](bf16)^T
//    256 threads = 8 waves; 128x128 block tile; wave w -> 16 rows x 128 cols
//    (8 accum frags, 16 WMMAs per 64-deep k-chunk).  k-step outer / n-tile
//    inner so 8 consecutive WMMAs share A (reuse_a hint); B-fragment loads
//    pipelined one n-tile ahead.
//    mode 0: write f32 C to outF[M,N]
//    mode 1: QKV epilogue -> q[h,n,d], k[h,n,d], vT[h,d,n] (bf16)
// ---------------------------------------------------------------------------
#define GLDA 72   // 64 + 8-elem (16B) pad: keeps b128 alignment, skews banks

__global__ void __launch_bounds__(256)
wmma_gemm_kernel(const bf16_t* __restrict__ A, const bf16_t* __restrict__ Bt,
                 int M, int N, int K,
                 float* __restrict__ outF,
                 bf16_t* __restrict__ qOut, bf16_t* __restrict__ kOut,
                 bf16_t* __restrict__ vtOut, int mode) {
  __shared__ __align__(16) bf16_t ldsA[2][128][GLDA];
  __shared__ __align__(16) bf16_t ldsB[2][128][GLDA];

  const int tid   = threadIdx.x;
  const int lane  = tid & 31;
  const int wave  = tid >> 5;                 // 0..7
  const int tileM = blockIdx.y * 128;
  const int tileN = blockIdx.x * 128;
  const int laneN = lane & 15;
  const int kOffA = (lane < 16) ? 0 : 16;     // A frag K split (contiguous k = 16h + s)
  const int kOffB = (lane < 16) ? 0 : 16;     // B frag K split

  auto issueTile = [&](int buf, int k0) {     // 128x64 A + 128x64 B, 16B chunks
    #pragma unroll
    for (int c = 0; c < 4; ++c) {
      const int chunk = tid + c * 256;        // 0..1023
      const int row = chunk >> 3, off = (chunk & 7) * 8;
      async_ld16(&ldsA[buf][row][off], A  + (size_t)(tileM + row) * K + k0 + off);
      async_ld16(&ldsB[buf][row][off], Bt + (size_t)(tileN + row) * K + k0 + off);
    }
  };

  v8f acc[8] = {};
  issueTile(0, 0);
  const int nk = K / 64;
  for (int i = 0; i < nk; ++i) {
    const int buf = i & 1;
    wait_async0();
    __syncthreads();
    if (i + 1 < nk) issueTile(buf ^ 1, (i + 1) * 64);   // overlap with WMMA
    // hoist both A fragments (k-steps 0,1)
    FragAB a[2];
    #pragma unroll
    for (int ks = 0; ks < 2; ++ks)
      load_frag(a[ks], &ldsA[buf][wave * 16 + laneN][ks * 32 + kOffA], 8);
    // k-step outer: 8 consecutive WMMAs share a[ks] -> reuse_a
    #pragma unroll
    for (int ks = 0; ks < 2; ++ks) {
      const bf16_t* bbase = &ldsB[buf][laneN][ks * 32 + kOffB];
      FragAB bc, bn;
      load_frag(bc, bbase, 8);                          // nt = 0
      load_frag(bn, bbase + 16 * GLDA, 8);              // prefetch nt = 1
      acc[0] = wmma_bf16<false>(a[ks].v, bc.v, acc[0]);
      bc = bn;
      #pragma unroll
      for (int nt = 1; nt < 8; ++nt) {
        if (nt < 7) load_frag(bn, bbase + (nt + 1) * 16 * GLDA, 8);
        acc[nt] = wmma_bf16<true>(a[ks].v, bc.v, acc[nt]);
        bc = bn;
      }
    }
    __syncthreads();
  }

  const int rBase = (lane >> 4) * 8;          // C/D: lanes 16-31 hold rows 8..15
  if (mode == 0) {
    #pragma unroll
    for (int nt = 0; nt < 8; ++nt) {
      const int col = tileN + nt * 16 + laneN;
      #pragma unroll
      for (int r = 0; r < 8; ++r)
        outF[(size_t)(tileM + wave * 16 + rBase + r) * N + col] = acc[nt][r];
    }
  } else {
    #pragma unroll
    for (int nt = 0; nt < 8; ++nt) {
      const int c     = tileN + nt * 16 + laneN;   // frag never crosses 2048
      const int which = c >> 11;
      const int cc    = c & (DIM - 1);
      const int h     = cc >> 7, d = cc & (DHEAD - 1);
      if (which < 2) {                             // q or k: [h, n, d] row-major
        bf16_t* dst = (which == 0 ? qOut : kOut) + (size_t)h * SEQ * DHEAD + d;
        #pragma unroll
        for (int r = 0; r < 8; ++r) {
          const int n = tileM + wave * 16 + rBase + r;
          dst[(size_t)n * DHEAD] = (bf16_t)acc[nt][r];
        }
      } else {                                     // v transposed: [h, d, n]
        uint32_t pw[4];
        #pragma unroll
        for (int p = 0; p < 4; ++p) {
          bf16_t lo = (bf16_t)acc[nt][2 * p], hi = (bf16_t)acc[nt][2 * p + 1];
          uint16_t l16, h16;
          __builtin_memcpy(&l16, &lo, 2); __builtin_memcpy(&h16, &hi, 2);
          pw[p] = (uint32_t)l16 | ((uint32_t)h16 << 16);
        }
        const int n0 = tileM + wave * 16 + rBase;
        uint4 pk = make_uint4(pw[0], pw[1], pw[2], pw[3]);
        *(uint4*)(vtOut + ((size_t)h * DHEAD + d) * SEQ + n0) = pk;
      }
    }
  }
}

// ---------------------------------------------------------------------------
// 4) Rotary on q,k (bf16 in-place), q also *= sqrt(DHEAD)
// ---------------------------------------------------------------------------
__global__ void __launch_bounds__(64)
rotary_kernel(bf16_t* __restrict__ q, bf16_t* __restrict__ k,
              const float* __restrict__ rot) {
  const int idx = blockIdx.x;             // h*SEQ + n
  const int n   = idx & (SEQ - 1);
  const int d   = threadIdx.x;            // pair (d, d+64); same angle
  const float ang = rot[(size_t)n * DHEAD + d];
  float c, s; __sincosf(ang, &s, &c);
  const size_t base = (size_t)idx * DHEAD;
  const float qs = 11.313708498984761f;   // sqrt(128)
  float q1 = (float)q[base + d], q2 = (float)q[base + d + 64];
  float k1 = (float)k[base + d], k2 = (float)k[base + d + 64];
  q[base + d]      = (bf16_t)((q1 * c - q2 * s) * qs);
  q[base + d + 64] = (bf16_t)((q2 * c + q1 * s) * qs);
  k[base + d]      = (bf16_t)(k1 * c - k2 * s);
  k[base + d + 64] = (bf16_t)(k2 * c + k1 * s);
}

// ---------------------------------------------------------------------------
// 5) Flash attention (causal).  4 waves x 16 query rows (64 q rows / block).
//    K & V^T tiles double-buffered in LDS via async copies shared by all
//    waves; online softmax; P re-laid out through wave-local LDS.  Compute
//    loops ordered for A-operand reuse; B-fragments pipelined one step.
// ---------------------------------------------------------------------------
#define KTLD 136   // 128 + 8 pad
#define VTLD 72    // 64 + 8 pad

__global__ void __launch_bounds__(128)
flash_attn_kernel(const bf16_t* __restrict__ Q, const bf16_t* __restrict__ Kb,
                  const bf16_t* __restrict__ Vt, bf16_t* __restrict__ O) {
  __shared__ __align__(16) bf16_t kTile[2][64][KTLD];   // [key][d]
  __shared__ __align__(16) bf16_t vTile[2][DHEAD][VTLD];// [d][key]
  __shared__ __align__(16) bf16_t pTile[4][16][64];     // wave-local P

  const int h    = blockIdx.y;
  const int qblk = blockIdx.x;
  const int tid  = threadIdx.x;
  const int lane = tid & 31;
  const int wave = tid >> 5;
  const int laneN = lane & 15;
  const int rBase = (lane >> 4) * 8;
  const int kOffA = (lane < 16) ? 0 : 16;
  const int kOffB = (lane < 16) ? 0 : 16;
  const int qRow0 = qblk * 64 + wave * 16;

  const bf16_t* Qh = Q  + (size_t)h * SEQ * DHEAD;
  const bf16_t* Kh = Kb + (size_t)h * SEQ * DHEAD;
  const bf16_t* Vh = Vt + (size_t)h * DHEAD * SEQ;

  auto issueKV = [&](int buf, int key0) {
    #pragma unroll
    for (int c = 0; c < 8; ++c) {
      const int chunk = tid + c * 128;                  // 0..1023
      { // K tile: 64 keys x 128 d (16 chunks/row)
        const int row = chunk >> 4, off = (chunk & 15) * 8;
        async_ld16(&kTile[buf][row][off], Kh + (size_t)(key0 + row) * DHEAD + off);
      }
      { // V^T tile: 128 d x 64 keys (8 chunks/row)
        const int row = chunk >> 3, off = (chunk & 7) * 8;
        async_ld16(&vTile[buf][row][off], Vh + (size_t)row * SEQ + key0 + off);
      }
    }
  };

  // Preload 16x128 Q tile as 4 A-fragments (per-wave, from global)
  FragAB qf[4];
  {
    const bf16_t* qp = Qh + (size_t)(qRow0 + laneN) * DHEAD;
    #pragma unroll
    for (int ks = 0; ks < 4; ++ks)
      load_frag(qf[ks], qp + ks * 32 + kOffA, 8);
  }

  v8f o[8] = {};
  float m[8], l[8];
  #pragma unroll
  for (int r = 0; r < 8; ++r) { m[r] = -1e30f; l[r] = 0.f; }

  issueKV(0, 0);
  for (int jb = 0; jb <= qblk; ++jb) {
    const int buf  = jb & 1;
    const int key0 = jb * 64;
    wait_async0();
    __syncthreads();
    if (jb < qblk) issueKV(buf ^ 1, key0 + 64);         // overlap with compute

    // ---- S = Q K^T: k-step outer so 4 WMMAs share qf[ks] (reuse_a) ----
    v8f s[4] = {};
    #pragma unroll
    for (int ks = 0; ks < 4; ++ks) {
      const bf16_t* bbase = &kTile[buf][laneN][ks * 32 + kOffB];
      FragAB bc, bn;
      load_frag(bc, bbase, 8);                          // nt = 0
      load_frag(bn, bbase + 16 * KTLD, 8);              // prefetch nt = 1
      s[0] = wmma_bf16<false>(qf[ks].v, bc.v, s[0]);
      bc = bn;
      #pragma unroll
      for (int nt = 1; nt < 4; ++nt) {
        if (nt < 3) load_frag(bn, bbase + (nt + 1) * 16 * KTLD, 8);
        s[nt] = wmma_bf16<true>(qf[ks].v, bc.v, s[nt]);
        bc = bn;
      }
    }
    if (jb == qblk) {                      // causal mask on diagonal tile
      #pragma unroll
      for (int nt = 0; nt < 4; ++nt) {
        const int key = key0 + nt * 16 + laneN;
        #pragma unroll
        for (int r = 0; r < 8; ++r)
          if (key > qRow0 + rBase + r) s[nt][r] = -1e30f;
      }
    }
    // ---- online softmax (row = 16 lanes of one half-wave) ----
    float pr[4][8], alpha[8];
    #pragma unroll
    for (int r = 0; r < 8; ++r) {
      float mx = fmaxf(fmaxf(s[0][r], s[1][r]), fmaxf(s[2][r], s[3][r]));
      #pragma unroll
      for (int msk = 1; msk < 16; msk <<= 1) mx = fmaxf(mx, __shfl_xor(mx, msk, 32));
      const float nm = fmaxf(m[r], mx);
      alpha[r] = __expf(m[r] - nm);
      m[r] = nm;
      float rs = 0.f;
      #pragma unroll
      for (int nt = 0; nt < 4; ++nt) {
        const float p = __expf(s[nt][r] - nm);
        pr[nt][r] = p; rs += p;
      }
      #pragma unroll
      for (int msk = 1; msk < 16; msk <<= 1) rs += __shfl_xor(rs, msk, 32);
      l[r] = l[r] * alpha[r] + rs;
    }
    #pragma unroll
    for (int dt = 0; dt < 8; ++dt)
      #pragma unroll
      for (int r = 0; r < 8; ++r) o[dt][r] *= alpha[r];
    // ---- stage P (C-layout f32 -> row-major bf16, wave-local LDS) ----
    #pragma unroll
    for (int nt = 0; nt < 4; ++nt)
      #pragma unroll
      for (int r = 0; r < 8; ++r)
        pTile[wave][rBase + r][nt * 16 + laneN] = (bf16_t)pr[nt][r];
    // ---- O += P @ V: k-step outer so 8 WMMAs share pa[ks] (reuse_a) ----
    FragAB pa[2];
    {
      const bf16_t* pp = &pTile[wave][laneN][0];
      #pragma unroll
      for (int ks = 0; ks < 2; ++ks)
        load_frag(pa[ks], pp + ks * 32 + kOffA, 8);
    }
    #pragma unroll
    for (int ks = 0; ks < 2; ++ks) {
      const bf16_t* vbase = &vTile[buf][laneN][ks * 32 + kOffB];
      FragAB bc, bn;
      load_frag(bc, vbase, 8);                          // dt = 0
      load_frag(bn, vbase + 16 * VTLD, 8);              // prefetch dt = 1
      o[0] = wmma_bf16<false>(pa[ks].v, bc.v, o[0]);
      bc = bn;
      #pragma unroll
      for (int dt = 1; dt < 8; ++dt) {
        if (dt < 7) load_frag(bn, vbase + (dt + 1) * 16 * VTLD, 8);
        o[dt] = wmma_bf16<true>(pa[ks].v, bc.v, o[dt]);
        bc = bn;
      }
    }
    __syncthreads();
  }
  // ---- epilogue: O/l -> bf16 [n, HEADS*DHEAD] ----
  #pragma unroll
  for (int r = 0; r < 8; ++r) {
    const float inv = 1.f / l[r];
    bf16_t* dst = O + (size_t)(qRow0 + rBase + r) * DIM + h * DHEAD;
    #pragma unroll
    for (int dt = 0; dt < 8; ++dt)
      dst[dt * 16 + laneN] = (bf16_t)(o[dt][r] * inv);
  }
}

// ---------------------------------------------------------------------------
// Launch
// ---------------------------------------------------------------------------
extern "C" void kernel_launch(void* const* d_in, const int* in_sizes, int n_in,
                              void* d_out, int out_size, void* d_ws, size_t ws_size,
                              hipStream_t stream) {
  const float* x     = (const float*)d_in[0];
  const float* rot   = (const float*)d_in[1];
  const float* gamma = (const float*)d_in[2];
  const float* w_qkv = (const float*)d_in[3];
  const float* w_out = (const float*)d_in[4];
  float* out = (float*)d_out;

  char* ws = (char*)d_ws;
  bf16_t* xn    = (bf16_t*)ws; ws += (size_t)SEQ * DIM * 2;          // 16 MB
  bf16_t* wqkvT = (bf16_t*)ws; ws += (size_t)NQKV * DIM * 2;         // 24 MB
  bf16_t* woutT = (bf16_t*)ws; ws += (size_t)DIM * DIM * 2;          //  8 MB
  bf16_t* qbuf  = (bf16_t*)ws; ws += (size_t)HEADS * SEQ * DHEAD * 2;// 16 MB
  bf16_t* kbuf  = (bf16_t*)ws; ws += (size_t)HEADS * SEQ * DHEAD * 2;// 16 MB
  bf16_t* vtbuf = (bf16_t*)ws; ws += (size_t)HEADS * SEQ * DHEAD * 2;// 16 MB
  bf16_t* aout  = (bf16_t*)ws; ws += (size_t)SEQ * DIM * 2;          // 16 MB

  rmsnorm_kernel<<<SEQ, 256, 0, stream>>>(x, gamma, xn);
  transpose_cast_kernel<<<dim3(NQKV / 32, DIM / 32), 256, 0, stream>>>(w_qkv, wqkvT, DIM, NQKV);
  transpose_cast_kernel<<<dim3(DIM / 32, DIM / 32), 256, 0, stream>>>(w_out, woutT, DIM, DIM);
  wmma_gemm_kernel<<<dim3(NQKV / 128, SEQ / 128), 256, 0, stream>>>(
      xn, wqkvT, SEQ, NQKV, DIM, nullptr, qbuf, kbuf, vtbuf, 1);
  rotary_kernel<<<HEADS * SEQ, 64, 0, stream>>>(qbuf, kbuf, rot);
  flash_attn_kernel<<<dim3(SEQ / 64, HEADS), 128, 0, stream>>>(qbuf, kbuf, vtbuf, aout);
  wmma_gemm_kernel<<<dim3(DIM / 128, SEQ / 128), 256, 0, stream>>>(
      aout, woutT, SEQ, DIM, DIM, out, nullptr, nullptr, nullptr, 0);
}
